// RainStreaks_16106127360108
// MI455X (gfx1250) — compile-verified, matrix operand order checked by reading, not run on hardware
//
#include <hip/hip_runtime.h>

typedef __attribute__((ext_vector_type(16))) _Float16 v16h;
typedef __attribute__((ext_vector_type(8)))  float    v8f;
typedef __attribute__((ext_vector_type(4)))  float    f4;

#define H 512
#define W 512
#define NSTREAK 40

// ---------------------------------------------------------------------------
// Kernel 1: rasterize the streak-overlap count map K[h,w] with WMMA and
// convert to f[h,w] = (1-a)^K in one pass. One wave (32 threads) computes one
// 16x16 tile of K via  K_tile = A(16x64 row-mask) x B(64x16 col-mask),
// done as two v_wmma_f32_16x16x32_f16 accumulations (streaks padded 40->64
// with zero row-masks). 0/1 f16 inputs and K<=40 make the count exact in f32.
// ---------------------------------------------------------------------------
__global__ __launch_bounds__(32) void RainStreaks_raster_wmma(
    const float* __restrict__ alpha,
    const int*   __restrict__ xc,
    const int*   __restrict__ y0,
    const int*   __restrict__ y1off,
    float*       __restrict__ fmap) {
  __shared__ int s_xc[64];
  __shared__ int s_y0[64];
  __shared__ int s_y1[64];

  const int lane = threadIdx.x;           // 0..31, wave32
  for (int s = lane; s < 64; s += 32) {
    if (s < NSTREAK) {
      s_xc[s] = xc[s];
      s_y0[s] = y0[s];
      s_y1[s] = y1off[s] + (H / 2);
    } else {                              // pad: empty row range -> A element 0
      s_xc[s] = 0;
      s_y0[s] = 0;
      s_y1[s] = 0;
    }
  }
  __syncthreads();

  const int half = lane >> 4;             // 0: lanes 0-15, 1: lanes 16-31
  const int m    = lane & 15;
  const int row  = blockIdx.y * 16 + m;   // A-matrix M index for this lane
  const int col  = blockIdx.x * 16 + m;   // B-matrix N index for this lane

  // Build A (row masks) and B (col masks) in the 16-bit 16x32 operand layout:
  // element e -> VGPR v=e/2; K = 16*(v/4) + 8*half + 2*(v%4) + (e&1).
  v16h a0 = {}, a1 = {}, b0 = {}, b1 = {};
#pragma unroll
  for (int e = 0; e < 16; ++e) {
    const int v  = e >> 1;
    const int kl = ((v >> 2) << 4) + (half << 3) + ((v & 3) << 1) + (e & 1);
#pragma unroll
    for (int chunk = 0; chunk < 2; ++chunk) {
      const int s  = kl + chunk * 32;     // streak index (>=40 are zero-padded)
      const int ylo = s_y0[s], yhi = s_y1[s];
      int c0 = s_xc[s] - 1; if (c0 < 0) c0 = 0;
      const int c1 = s_xc[s] + 1;
      const _Float16 av = (row >= ylo && row < yhi) ? (_Float16)1.0f : (_Float16)0.0f;
      const _Float16 bv = (col >= c0  && col < c1 ) ? (_Float16)1.0f : (_Float16)0.0f;
      if (chunk == 0) { a0[e] = av; b0[e] = bv; }
      else            { a1[e] = av; b1[e] = bv; }
    }
  }

  v8f acc = {};
  // (neg_a, A, neg_b, B, c_mod, C, reuse_a, reuse_b)
  acc = __builtin_amdgcn_wmma_f32_16x16x32_f16(false, a0, false, b0,
                                               (short)0, acc, false, false);
  acc = __builtin_amdgcn_wmma_f32_16x16x32_f16(false, a1, false, b1,
                                               (short)0, acc, false, false);

  const float a  = alpha[0];
  const float l2 = log2f(1.0f - a);       // f = (1-a)^K = exp2(K*log2(1-a))

  // C/D layout: VGPR v of lane -> element (M = v + 8*half, N = lane%16)
#pragma unroll
  for (int v = 0; v < 8; ++v) {
    const int M = blockIdx.y * 16 + v + (half << 3);
    const int N = blockIdx.x * 16 + m;
    const float K = acc[v];
    const float f = (K == 0.0f) ? 1.0f : exp2f(K * l2);  // exact passthrough when K==0
    fmap[M * W + N] = f;
  }
}

// ---------------------------------------------------------------------------
// Kernel 2: bandwidth-bound blend. 201 MB of streamed traffic -> NT hints on
// x/out; the 1 MB f-map is loaded with default (RT) policy so it stays hot in
// L2 across the 96 B*C planes. One float4 per thread, fully coalesced B128.
// ---------------------------------------------------------------------------
__global__ __launch_bounds__(256) void RainStreaks_blend(
    const float* __restrict__ x,
    const float* __restrict__ fmap,
    float*       __restrict__ out,
    int total4) {
  const int i = blockIdx.x * 256 + threadIdx.x;
  if (i >= total4) return;

  const f4* __restrict__ xv4 = (const f4*)x;
  const f4* __restrict__ fv4 = (const f4*)fmap;
  f4*       __restrict__ ov4 = (f4*)out;

  const int hw4 = i & ((H * W / 4) - 1);  // 65536 float4 per plane (pow2)

  const f4 xv = __builtin_nontemporal_load(&xv4[i]);
  const f4 fv = fv4[hw4];

  f4 r;
  r.x = fminf(fmaxf(xv.x * fv.x + (1.0f - fv.x), 0.0f), 1.0f);
  r.y = fminf(fmaxf(xv.y * fv.y + (1.0f - fv.y), 0.0f), 1.0f);
  r.z = fminf(fmaxf(xv.z * fv.z + (1.0f - fv.z), 0.0f), 1.0f);
  r.w = fminf(fmaxf(xv.w * fv.w + (1.0f - fv.w), 0.0f), 1.0f);

  __builtin_nontemporal_store(r, &ov4[i]);
}

extern "C" void kernel_launch(void* const* d_in, const int* in_sizes, int n_in,
                              void* d_out, int out_size, void* d_ws, size_t ws_size,
                              hipStream_t stream) {
  const float* x     = (const float*)d_in[0];   // [32,3,512,512] f32
  const float* alpha = (const float*)d_in[1];   // [1] f32
  const int*   xc    = (const int*)  d_in[2];   // [40] i32
  const int*   y0    = (const int*)  d_in[3];   // [40] i32
  const int*   y1off = (const int*)  d_in[4];   // [40] i32
  float*       out   = (float*)d_out;
  float*       fmap  = (float*)d_ws;            // 512*512*4 = 1 MB scratch

  // Pass 1: WMMA rasterization of the (1-a)^K map, one wave per 16x16 tile.
  dim3 g1(W / 16, H / 16);                      // 32 x 32 tiles
  RainStreaks_raster_wmma<<<g1, 32, 0, stream>>>(alpha, xc, y0, y1off, fmap);

  // Pass 2: streaming blend over all B*C*H*W elements.
  const int total4 = in_sizes[0] / 4;           // 6,291,456 float4
  const int blocks = (total4 + 255) / 256;      // 24,576 blocks
  RainStreaks_blend<<<blocks, 256, 0, stream>>>(x, fmap, out, total4);
}
